// SelfAttention_8486855377136
// MI455X (gfx1250) — matrix operand order repaired and checked by reading, not compiled, not run on hardware
//
#include <hip/hip_runtime.h>
#include <hip/hip_bf16.h>

// ---------------------------------------------------------------------------
// MI455X (gfx1250) fused self-attention:
//   1) QKV projections : f32 x @ f32 W -> bf16 Q/K/V in [B,H,S,Hd] (WMMA bf16)
//   2) flash attention : fused QK^T / online-softmax / PV, never spills SxS
//   3) output proj     : bf16 attn @ f32 Wo -> f32 out          (WMMA bf16)
// Matrix work on v_wmma_f32_16x16x32_bf16; tile staging uses the Tensor Data
// Mover (tensor_load_to_lds + s_wait_tensorcnt) where no conversion is needed,
// and global_prefetch for the next tiles.
// ---------------------------------------------------------------------------

typedef __bf16 bf16_t;
typedef __bf16 v16bf __attribute__((ext_vector_type(16)));
typedef float  v8f   __attribute__((ext_vector_type(8)));
typedef unsigned int u32x4 __attribute__((ext_vector_type(4)));
typedef int          i32x4 __attribute__((ext_vector_type(4)));
typedef int          i32x8 __attribute__((ext_vector_type(8)));

#define EMBED    1024
#define HEADS    16
#define HEAD_DIM 64
#define BATCH    4
#define SEQ      2048
#define ROWS     (BATCH * SEQ)   // 8192

// ---- pack two f32 -> dword of 2 bf16 ---------------------------------------
static __device__ __forceinline__ unsigned pack2bf(float x, float y) {
  union { bf16_t h[2]; unsigned u; } p;
  p.h[0] = (bf16_t)x; p.h[1] = (bf16_t)y;
  return p.u;
}

// ---- Tensor Data Mover: 2D bf16 tile (global -> LDS), per CDNA5 ISA -----
// tile_d0 elems contiguous (dim0), tile_d1 lines, line stride = stride elems.
static __device__ __forceinline__ void tdm_load_2d(unsigned lds_addr,
                                                   const void* gaddr,
                                                   unsigned tile_d0,
                                                   unsigned tile_d1,
                                                   unsigned stride) {
  unsigned long long ga = (unsigned long long)gaddr;
  u32x4 g0;
  g0[0] = 1u;                                    // count=1 (valid), user mode
  g0[1] = lds_addr;                              // LDS byte address
  g0[2] = (unsigned)(ga & 0xffffffffu);          // global_addr[31:0]
  g0[3] = (unsigned)((ga >> 32) & 0x1ffffffu) | (2u << 30);  // [56:32] | type=2
  i32x8 g1;
  g1[0] = (int)(1u << 16);                       // wg_mask=0, data_size=1 (2B)
  g1[1] = (int)(tile_d0 << 16);                  // tensor_dim0[15:0] @ [63:48]
  g1[2] = (int)((tile_d0 >> 16) | (tile_d1 << 16)); // dim0 hi | tensor_dim1 lo
  g1[3] = (int)((tile_d1 >> 16) | (tile_d0 << 16)); // dim1 hi | tile_dim0
  g1[4] = (int)(tile_d1 & 0xffffu);              // tile_dim1, tile_dim2=0
  g1[5] = (int)stride;                           // tensor_dim0_stride[31:0]
  g1[6] = 0;                                     // stride hi, dim1_stride=0
  g1[7] = 0;
  i32x4 z4 = {0, 0, 0, 0};
#if __clang_major__ >= 23
  i32x8 z8 = {0, 0, 0, 0, 0, 0, 0, 0};
  __builtin_amdgcn_tensor_load_to_lds(g0, g1, z4, z4, z8, 0);
#else
  __builtin_amdgcn_tensor_load_to_lds(g0, g1, z4, z4, 0);
#endif
}

// ---- WMMA fragment builders (layouts per CDNA5 ISA 7.12.2) ------
static __device__ __forceinline__ v16bf make_afrag(const bf16_t* tile, int ld,
                                                   int m, int half, int kbase) {
  v16bf a;
  unsigned* ap = (unsigned*)&a;
  const unsigned* row = (const unsigned*)(tile + (size_t)m * ld);
#pragma unroll
  for (int v = 0; v < 8; ++v) {
    int k = kbase + ((v < 4) ? (half * 8 + v * 2) : (16 + half * 8 + (v - 4) * 2));
    ap[v] = row[k >> 1];
  }
  return a;
}

static __device__ __forceinline__ v16bf make_bfrag(const bf16_t* tile, int ld,
                                                   int n, int half, int kbase) {
  v16bf b;
  unsigned* bp = (unsigned*)&b;
  const unsigned* col = (const unsigned*)(tile + (size_t)n * ld);
#pragma unroll
  for (int v = 0; v < 8; ++v)
    bp[v] = col[(kbase >> 1) + ((v < 4) ? (half * 4 + v) : (8 + half * 4 + (v - 4)))];
  return b;
}

// ---------------------------------------------------------------------------
// GEMM: out[M,N] = A[M,1024] @ W[1024,N] + bias    (M=8192, N per launch)
// Block: 256 thr (8 waves) -> 128(M) x 64(N) tile, BK=64 (8 WMMA / barrier).
// AMODE 0: A is f32 (x, converted while staging).
// AMODE 1: A is bf16 (attn out) -> staged by the Tensor Data Mover.
// OMODE 0: bf16 out in [B,H,S,Hd] head layout (QKV).
// OMODE 1: f32 out row-major [M,N] (final projection -> d_out).
// ---------------------------------------------------------------------------
template <int AMODE, int OMODE>
__global__ __launch_bounds__(256) void gemm_wmma(const void* __restrict__ Aptr,
                                                 const float* __restrict__ W,
                                                 const float* __restrict__ bias,
                                                 void* __restrict__ outp) {
  __shared__ bf16_t sA[128 * 64];  // [row][k]  k-contiguous (16 KB)
  __shared__ bf16_t sB[64 * 64];   // [n][k]    k-contiguous ( 8 KB)

  const int tid  = threadIdx.x;
  const int lane = tid & 31;
  const int wave = tid >> 5;
  const int m16  = lane & 15;
  const int half = lane >> 4;
  const int row0 = blockIdx.x * 128;
  const int col0 = blockIdx.y * 64;

  v8f acc[4];
#pragma unroll
  for (int nt = 0; nt < 4; ++nt)
#pragma unroll
    for (int i = 0; i < 8; ++i) acc[nt][i] = 0.0f;

  for (int k0 = 0; k0 < EMBED; k0 += 64) {
    // ---- stage A tile (128 x 64) bf16, k-minor ----
    if (AMODE == 0) {
      const int r   = tid >> 1;
      const int seg = (tid & 1) * 32;
      const float4* s4 =
          (const float4*)((const float*)Aptr + (size_t)(row0 + r) * EMBED + k0 + seg);
      unsigned* drow = (unsigned*)(sA + r * 64 + seg);
#pragma unroll
      for (int i = 0; i < 8; ++i) {
        float4 v = s4[i];
        drow[2 * i + 0] = pack2bf(v.x, v.y);
        drow[2 * i + 1] = pack2bf(v.z, v.w);
      }
    } else {
      if (wave == 0)  // scalar-uniform branch: single TDM issue per block
        tdm_load_2d((unsigned)(unsigned long long)(void*)sA,
                    (const bf16_t*)Aptr + (size_t)row0 * EMBED + k0,
                    /*tile_d0=*/64, /*tile_d1=*/128, /*stride=*/EMBED);
    }
    // ---- stage B tile (64k x 64n) transposed -> sB[n][k], packed dwords ----
    {
      const int kp   = tid >> 3;        // k pair id 0..31
      const int nseg = (tid & 7) * 8;   // 0..56
      const float* w0 = W + (size_t)(k0 + 2 * kp) * EMBED + col0 + nseg;
      const float* w1 = w0 + EMBED;
#pragma unroll
      for (int i = 0; i < 8; ++i)
        *(unsigned*)(sB + (nseg + i) * 64 + 2 * kp) = pack2bf(w0[i], w1[i]);
    }
    // ---- prefetch next K-step tiles into cache ----
    if (k0 + 64 < EMBED) {
      __builtin_prefetch(W + (size_t)(k0 + 64 + (tid & 63)) * EMBED + col0 +
                             (tid >> 6) * 16, 0, 1);
      if (AMODE == 0)
        __builtin_prefetch((const float*)Aptr + (size_t)(row0 + (tid & 127)) * EMBED +
                               k0 + 64 + (tid >> 7) * 32, 0, 1);
      else
        __builtin_prefetch((const bf16_t*)Aptr + (size_t)(row0 + (tid & 127)) * EMBED +
                               k0 + 64, 0, 1);
    }
    if (AMODE == 1 && wave == 0) __builtin_amdgcn_s_wait_tensorcnt(0);
    __syncthreads();

#pragma unroll
    for (int kc = 0; kc < 2; ++kc) {
      v16bf a = make_afrag(sA, 64, wave * 16 + m16, half, kc * 32);
#pragma unroll
      for (int nt = 0; nt < 4; ++nt) {
        v16bf b = make_bfrag(sB, 64, nt * 16 + m16, half, kc * 32);
        acc[nt] = __builtin_amdgcn_wmma_f32_16x16x32_bf16(
            false, a, false, b, (short)0, acc[nt], false, false);
      }
    }
    __syncthreads();
  }

  // ---- epilogue: bias add + store ----
#pragma unroll
  for (int nt = 0; nt < 4; ++nt) {
    const int col = col0 + nt * 16 + m16;
    const float bv = bias[col];
#pragma unroll
    for (int r = 0; r < 8; ++r) {
      const float val  = acc[nt][r] + bv;
      const int   grow = row0 + wave * 16 + r + 8 * half;
      if (OMODE == 0) {
        const int b = grow >> 11, s = grow & (SEQ - 1);
        const int h = col >> 6, hd = col & 63;
        bf16_t* out = (bf16_t*)outp;
        out[(((size_t)(b * HEADS + h) * SEQ + s) << 6) + hd] = (bf16_t)val;
      } else {
        float* out = (float*)outp;
        out[(size_t)grow * EMBED + col] = val;
      }
    }
  }
}

// ---------------------------------------------------------------------------
// Flash attention. Block = 256 thr (8 waves) handles 128 query rows of one
// (b,h). Loops over 64-key blocks; K tile staged by TDM, V tile transposed
// manually (TDM cannot transpose); online softmax.  grid = (SEQ/128, B*H)
// ---------------------------------------------------------------------------
__global__ __launch_bounds__(256) void attention_wmma(const bf16_t* __restrict__ Q,
                                                      const bf16_t* __restrict__ K,
                                                      const bf16_t* __restrict__ V,
                                                      bf16_t* __restrict__ O) {
  __shared__ bf16_t sK[64 * 64];       // [key][hd]   (natural, hd-contiguous)
  __shared__ bf16_t sVt[64 * 64];      // [hd][key]   (transposed)
  __shared__ bf16_t sP[8][16 * 64];    // per-wave P tile [qrow][key]

  const int tid  = threadIdx.x;
  const int lane = tid & 31;
  const int wave = tid >> 5;
  const int m16  = lane & 15;
  const int half = lane >> 4;
  const int bh   = blockIdx.y;           // b*HEADS + h
  const int q0   = blockIdx.x * 128;

  const bf16_t* Qb = Q + (size_t)bh * SEQ * HEAD_DIM;
  const bf16_t* Kb = K + (size_t)bh * SEQ * HEAD_DIM;
  const bf16_t* Vb = V + (size_t)bh * SEQ * HEAD_DIM;

  // Q fragments for this wave's 16 rows, loaded once (A layout, Hd-contiguous)
  v16bf qf[2];
  {
    const unsigned* qrow =
        (const unsigned*)(Qb + (size_t)(q0 + wave * 16 + m16) * HEAD_DIM);
#pragma unroll
    for (int kc = 0; kc < 2; ++kc) {
      unsigned* ap = (unsigned*)&qf[kc];
#pragma unroll
      for (int v = 0; v < 8; ++v) {
        int k = kc * 32 +
                ((v < 4) ? (half * 8 + v * 2) : (16 + half * 8 + (v - 4) * 2));
        ap[v] = qrow[k >> 1];
      }
    }
  }

  v8f acc[4];
  float mst[8], lst[8];
#pragma unroll
  for (int ht = 0; ht < 4; ++ht)
#pragma unroll
    for (int i = 0; i < 8; ++i) acc[ht][i] = 0.0f;
#pragma unroll
  for (int r = 0; r < 8; ++r) { mst[r] = -3.0e38f; lst[r] = 0.0f; }

  const float scale = 0.125f;  // 1/sqrt(64)

  for (int j0 = 0; j0 < SEQ; j0 += 64) {
    // ---- K tile via Tensor Data Mover (no conversion, natural layout) ----
    if (wave == 0)
      tdm_load_2d((unsigned)(unsigned long long)(void*)sK,
                  Kb + (size_t)j0 * HEAD_DIM,
                  /*tile_d0=*/64, /*tile_d1=*/64, /*stride=*/64);
    // ---- V tile transposed: sVt[hd][key], packed dword stores ----
    {
      const int kp = tid >> 3, part = (tid & 7) * 8;
      const unsigned short* a =
          (const unsigned short*)(Vb + (size_t)(j0 + 2 * kp) * 64 + part);
      const unsigned short* b =
          (const unsigned short*)(Vb + (size_t)(j0 + 2 * kp + 1) * 64 + part);
#pragma unroll
      for (int i = 0; i < 8; ++i)
        *(unsigned*)(sVt + (part + i) * 64 + 2 * kp) =
            (unsigned)a[i] | ((unsigned)b[i] << 16);
    }
    // ---- prefetch next key block ----
    if (j0 + 64 < SEQ) {
      __builtin_prefetch(Kb + (size_t)(j0 + 64) * 64 + tid * 16, 0, 1);
      __builtin_prefetch(Vb + (size_t)(j0 + 64) * 64 + tid * 16, 0, 1);
    }
    if (wave == 0) __builtin_amdgcn_s_wait_tensorcnt(0);
    __syncthreads();

    // ---- scores S = Q @ K^T (16 x 64), 8 WMMA ----
    v8f sc[4];
#pragma unroll
    for (int nt = 0; nt < 4; ++nt) {
#pragma unroll
      for (int i = 0; i < 8; ++i) sc[nt][i] = 0.0f;
#pragma unroll
      for (int kc = 0; kc < 2; ++kc) {
        v16bf bk = make_bfrag(sK, 64, nt * 16 + m16, half, kc * 32);
        sc[nt] = __builtin_amdgcn_wmma_f32_16x16x32_bf16(
            false, qf[kc], false, bk, (short)0, sc[nt], false, false);
      }
#pragma unroll
      for (int i = 0; i < 8; ++i) sc[nt][i] *= scale;
    }

    // ---- online softmax (row stats per lane-half, 8 rows each) ----
    float alpha[8];
#pragma unroll
    for (int r = 0; r < 8; ++r) {
      float t = fmaxf(fmaxf(sc[0][r], sc[1][r]), fmaxf(sc[2][r], sc[3][r]));
#pragma unroll
      for (int mask = 1; mask < 16; mask <<= 1)
        t = fmaxf(t, __shfl_xor(t, mask, 32));
      const float mn = fmaxf(mst[r], t);
      alpha[r] = __expf(mst[r] - mn);
      mst[r]   = mn;
    }
    float rsum[8];
#pragma unroll
    for (int r = 0; r < 8; ++r) rsum[r] = 0.0f;
#pragma unroll
    for (int nt = 0; nt < 4; ++nt)
#pragma unroll
      for (int r = 0; r < 8; ++r) {
        const float p = __expf(sc[nt][r] - mst[r]);
        sc[nt][r] = p;
        rsum[r] += p;
      }
#pragma unroll
    for (int r = 0; r < 8; ++r) {
      float t = rsum[r];
#pragma unroll
      for (int mask = 1; mask < 16; mask <<= 1) t += __shfl_xor(t, mask, 32);
      lst[r] = lst[r] * alpha[r] + t;
#pragma unroll
      for (int ht = 0; ht < 4; ++ht) acc[ht][r] *= alpha[r];
    }

    // ---- P: C-layout f32 -> A-layout bf16 via wave-private LDS ----
    bf16_t* pw = sP[wave];
#pragma unroll
    for (int nt = 0; nt < 4; ++nt)
#pragma unroll
      for (int r = 0; r < 8; ++r)
        pw[(r + 8 * half) * 64 + nt * 16 + m16] = (bf16_t)sc[nt][r];
    asm volatile("s_wait_dscnt 0" ::: "memory");  // wave-private: no barrier

    v16bf pf[2];
#pragma unroll
    for (int kc2 = 0; kc2 < 2; ++kc2)
      pf[kc2] = make_afrag(pw, 64, m16, half, kc2 * 32);

    // ---- acc += P @ V (16 x 64), 8 WMMA ----
#pragma unroll
    for (int ht = 0; ht < 4; ++ht)
#pragma unroll
      for (int kc2 = 0; kc2 < 2; ++kc2) {
        v16bf bv = make_bfrag(sVt, 64, ht * 16 + m16, half, kc2 * 32);
        acc[ht] = __builtin_amdgcn_wmma_f32_16x16x32_bf16(
            false, pf[kc2], false, bv, (short)0, acc[ht], false, false);
      }
    __syncthreads();
  }

  // ---- normalize + store attn output as bf16 [B,S,D] ----
  const int b = bh >> 4, h = bh & 15;
#pragma unroll
  for (int r = 0; r < 8; ++r) {
    const float inv = 1.0f / lst[r];
    const int s = q0 + wave * 16 + r + 8 * half;
#pragma unroll
    for (int ht = 0; ht < 4; ++ht) {
      const int d = h * 64 + ht * 16 + m16;
      O[(size_t)(b * SEQ + s) * EMBED + d] = (bf16_t)(acc[ht][r] * inv);
    }
  }
}

// ---------------------------------------------------------------------------
extern "C" void kernel_launch(void* const* d_in, const int* in_sizes, int n_in,
                              void* d_out, int out_size, void* d_ws, size_t ws_size,
                              hipStream_t stream) {
  (void)in_sizes; (void)n_in; (void)out_size; (void)ws_size;
  const float* x  = (const float*)d_in[0];
  const float* wq = (const float*)d_in[1];
  const float* bq = (const float*)d_in[2];
  const float* wk = (const float*)d_in[3];
  const float* bk = (const float*)d_in[4];
  const float* wv = (const float*)d_in[5];
  const float* bv = (const float*)d_in[6];
  const float* wo = (const float*)d_in[7];
  const float* bo = (const float*)d_in[8];

  // workspace: Q,K,V (head layout) + attn out, all bf16 (4 x 16 MB = 64 MB)
  bf16_t* Q = (bf16_t*)d_ws;
  bf16_t* K = Q + (size_t)ROWS * EMBED;
  bf16_t* V = K + (size_t)ROWS * EMBED;
  bf16_t* A = V + (size_t)ROWS * EMBED;

  dim3 gproj(ROWS / 128, EMBED / 64);          // 64 x 16 blocks
  gemm_wmma<0, 0><<<gproj, 256, 0, stream>>>(x, wq, bq, Q);
  gemm_wmma<0, 0><<<gproj, 256, 0, stream>>>(x, wk, bk, K);
  gemm_wmma<0, 0><<<gproj, 256, 0, stream>>>(x, wv, bv, V);

  dim3 gattn(SEQ / 128, BATCH * HEADS);        // 16 x 64 blocks
  attention_wmma<<<gattn, 256, 0, stream>>>(Q, K, V, A);

  gemm_wmma<1, 1><<<gproj, 256, 0, stream>>>(A, wo, bo, d_out);
}
